// MambaBlock_1769526526623
// MI455X (gfx1250) — matrix operand order repaired and checked by reading, not compiled, not run on hardware
//
#include <hip/hip_runtime.h>
#include <cstdint>

typedef unsigned short u16;
typedef unsigned int   u32;
typedef unsigned long long u64;

typedef __attribute__((ext_vector_type(16))) __bf16 v16bf;
typedef __attribute__((ext_vector_type(8)))  float  v8f;
typedef __attribute__((ext_vector_type(4)))  unsigned int ui4;
typedef __attribute__((ext_vector_type(8)))  int i8v;
typedef __attribute__((ext_vector_type(4)))  int i4v;

union Frag { v16bf v; uint4 q[2]; };

__device__ __forceinline__ u16 f2bf(float f) {
  u32 u = __float_as_uint(f);
  u32 r = (u + 0x7FFFu + ((u >> 16) & 1u)) >> 16;   // round-to-nearest-even
  return (u16)r;
}
__device__ __forceinline__ float bf2f(u16 h) {
  return __uint_as_float(((u32)h) << 16);
}

// ---------------------------------------------------------------- TDM 2D tile load (bf16 elements)
// Loads a tile0 x tile1 tile (elements, row-major, row stride = stride0 elements)
// from global gptr into LDS at byte offset lds_off. Rows >= td1 are zero-filled
// by the TDM's OOB handling. data_size = 2 bytes.
__device__ __forceinline__ void tdm_load_2d(const void* gptr, u32 lds_off,
                                            u32 td0, int td1_i, u32 stride0,
                                            u32 tile0, u32 tile1)
{
  u32 td1 = (td1_i < 0) ? 0u : (u32)td1_i;
  u64 ga = (u64)gptr;
  ui4 g0;
  g0.x = 1u;                                              // count=1, user mode
  g0.y = lds_off;                                         // lds_addr
  g0.z = (u32)(ga & 0xFFFFFFFFu);                         // global_addr[31:0]
  g0.w = (u32)((ga >> 32) & 0x1FFFFFFu) | (2u << 30);     // global_addr[56:32] | type=2
  i8v g1;
  g1[0] = (int)(1u << 16);                                // workgroup_mask=0, data_size=1 (2B)
  g1[1] = (int)((td0 & 0xFFFFu) << 16);                   // tensor_dim0[15:0]
  g1[2] = (int)(((td0 >> 16) & 0xFFFFu) | ((td1 & 0xFFFFu) << 16));   // td0[31:16] | td1[15:0]
  g1[3] = (int)(((td1 >> 16) & 0xFFFFu) | ((tile0 & 0xFFFFu) << 16)); // td1[31:16] | tile_dim0
  g1[4] = (int)(tile1 & 0xFFFFu);                         // tile_dim1 | tile_dim2=0
  g1[5] = (int)stride0;                                   // tensor_dim0_stride[31:0]
  g1[6] = 0;                                              // stride0[47:32]=0 | stride1[15:0]=0
  g1[7] = 0;
  i4v z4 = {0, 0, 0, 0};
#if defined(__clang_major__) && (__clang_major__ >= 23)
  i8v z8 = {0, 0, 0, 0, 0, 0, 0, 0};
  __builtin_amdgcn_tensor_load_to_lds(g0, g1, z4, z4, z8, 0);
#else
  __builtin_amdgcn_tensor_load_to_lds(g0, g1, z4, z4, 0);
#endif
}

// ---------------------------------------------------------------- converters
__global__ void cvt_bf16_kernel(const float* __restrict__ in, u16* __restrict__ out, size_t n) {
  size_t i  = (size_t)blockIdx.x * blockDim.x + threadIdx.x;
  size_t st = (size_t)gridDim.x * blockDim.x;
  for (; i < n; i += st) out[i] = f2bf(in[i]);
}

// ---------------------------------------------------------------- 16x16 inverse (Gauss-Jordan, partial pivot)
__global__ void invertA_kernel(const float* __restrict__ A, float* __restrict__ Ainv) {
  __shared__ float Msh[16][16];
  __shared__ float Ish[16][16];
  if (threadIdx.x == 0) {
    for (int r = 0; r < 16; ++r)
      for (int c = 0; c < 16; ++c) { Msh[r][c] = A[r * 16 + c]; Ish[r][c] = (r == c) ? 1.f : 0.f; }
    for (int col = 0; col < 16; ++col) {
      int p = col; float best = fabsf(Msh[col][col]);
      for (int r = col + 1; r < 16; ++r) { float v = fabsf(Msh[r][col]); if (v > best) { best = v; p = r; } }
      if (p != col) {
        for (int c = 0; c < 16; ++c) {
          float t = Msh[col][c]; Msh[col][c] = Msh[p][c]; Msh[p][c] = t;
          t = Ish[col][c]; Ish[col][c] = Ish[p][c]; Ish[p][c] = t;
        }
      }
      float ip = 1.f / Msh[col][col];
      for (int c = 0; c < 16; ++c) { Msh[col][c] *= ip; Ish[col][c] *= ip; }
      for (int r = 0; r < 16; ++r) {
        if (r == col) continue;
        float f = Msh[r][col];
        for (int c = 0; c < 16; ++c) { Msh[r][c] -= f * Msh[col][c]; Ish[r][c] -= f * Ish[col][c]; }
      }
    }
    for (int r = 0; r < 16; ++r)
      for (int c = 0; c < 16; ++c) Ainv[r * 16 + c] = Ish[r][c];
  }
}

// ---------------------------------------------------------------- WMMA GEMM with TDM-staged tiles
// C[m,n] = sum_k A[m,k] * B[n,k] + bias[n]
// mode 0: write f32 to Cf
// mode 1: s = silu(v); n <  halfN -> Ubf[m, n]        (bf16)
//                      n >= halfN -> Gbf[m, n-halfN]  (bf16)
#define BM 128
#define BN 64
#define BK 32

__global__ __launch_bounds__(256)
void gemm_bf16_wmma(const u16* __restrict__ Ag, const u16* __restrict__ Bg,
                    const float* __restrict__ bias,
                    float* __restrict__ Cf, u16* __restrict__ Ubf, u16* __restrict__ Gbf,
                    int M, int N, int K, int mode, int halfN)
{
  __shared__ u16 As[2][BM * BK];   // 2 x 8 KB
  __shared__ u16 Bs[2][BN * BK];   // 2 x 4 KB
  const int tid  = threadIdx.x;
  const int lane = tid & 31;
  const int wave = tid >> 5;    // 8 waves
  const int wm = wave >> 1;     // 0..3  (32-row strip)
  const int wn = wave & 1;      // 0..1  (32-col strip)
  const int tileM = blockIdx.y * BM;
  const int tileN = blockIdx.x * BN;
  const int nIter = K / BK;

  v8f acc[2][2];
#pragma unroll
  for (int a = 0; a < 2; ++a)
#pragma unroll
    for (int b = 0; b < 2; ++b)
#pragma unroll
      for (int r = 0; r < 8; ++r) acc[a][b][r] = 0.0f;

  // Double-buffered TDM pipeline: wave 0 drives the Tensor Data Mover.
  if (wave == 0) {
    tdm_load_2d(Ag + (size_t)tileM * K, (u32)(uintptr_t)&As[0][0],
                (u32)K, M - tileM, (u32)K, BK, BM);
    tdm_load_2d(Bg + (size_t)tileN * K, (u32)(uintptr_t)&Bs[0][0],
                (u32)K, N - tileN, (u32)K, BK, BN);
  }

  for (int it = 0; it < nIter; ++it) {
    const int buf = it & 1;
    if (wave == 0) {
      if (it + 1 < nIter) {
        const int k1 = (it + 1) * BK;
        const int nb = (it + 1) & 1;
        tdm_load_2d(Ag + (size_t)tileM * K + k1, (u32)(uintptr_t)&As[nb][0],
                    (u32)K, M - tileM, (u32)K, BK, BM);
        tdm_load_2d(Bg + (size_t)tileN * K + k1, (u32)(uintptr_t)&Bs[nb][0],
                    (u32)K, N - tileN, (u32)K, BK, BN);
        __builtin_amdgcn_s_wait_tensorcnt(2);   // in-order: oldest stage complete
      } else {
        __builtin_amdgcn_s_wait_tensorcnt(0);
      }
    }
    __syncthreads();

    // Fragments per documented 16-bit layouts:
    // A 16x32: lane<16 -> K[0..7],K[16..23]; lane>=16 -> K[8..15],K[24..31]; M = lane%16
    // B 32x16: lane<16 -> K[0..7],K[16..23]; lane>=16 -> K[8..15],K[24..31]; N = lane%16
    Frag afr[2], bfr[2];
    const int mrow = lane & 15;
    const int koff = (lane >> 4) << 3;
#pragma unroll
    for (int tm = 0; tm < 2; ++tm) {
      int m = wm * 32 + tm * 16 + mrow;
      afr[tm].q[0] = *(const uint4*)(&As[buf][m * BK + koff]);
      afr[tm].q[1] = *(const uint4*)(&As[buf][m * BK + 16 + koff]);
    }
    const int kk = lane & 16;
#pragma unroll
    for (int tn = 0; tn < 2; ++tn) {
      int n = wn * 32 + tn * 16 + mrow;
      const uint4* p = (const uint4*)(&Bs[buf][n * BK + koff]);
      bfr[tn].q[0] = p[0];
      bfr[tn].q[1] = p[2];
    }
#pragma unroll
    for (int tm = 0; tm < 2; ++tm)
#pragma unroll
      for (int tn = 0; tn < 2; ++tn)
        acc[tm][tn] = __builtin_amdgcn_wmma_f32_16x16x32_bf16(
            false, afr[tm].v, false, bfr[tn].v, (short)0, acc[tm][tn], false, false);
    __syncthreads();
  }

  // Epilogue. C layout: VGPR r -> M = r + (lane>=16 ? 8 : 0), N = lane%16
  const int mbase = (lane >> 4) << 3;
#pragma unroll
  for (int tm = 0; tm < 2; ++tm)
#pragma unroll
    for (int tn = 0; tn < 2; ++tn)
#pragma unroll
      for (int r = 0; r < 8; ++r) {
        int gm = tileM + wm * 32 + tm * 16 + mbase + r;
        int gn = tileN + wn * 32 + tn * 16 + (lane & 15);
        if (gm < M && gn < N) {
          float v = acc[tm][tn][r] + bias[gn];
          if (mode == 0) {
            Cf[(size_t)gm * N + gn] = v;
          } else {
            float s = v / (1.0f + __expf(-v));     // SiLU
            u16 h = f2bf(s);
            if (gn < halfN) Ubf[(size_t)gm * halfN + gn] = h;
            else            Gbf[(size_t)gm * halfN + (gn - halfN)] = h;
          }
        }
      }
}

// ---------------------------------------------------------------- selective scan (one wave32 per batch)
__global__ void scan_kernel(const float* __restrict__ Bm, const float* __restrict__ Cm,
                            const float* __restrict__ Dr, const float* __restrict__ A,
                            const float* __restrict__ Ainv, float* __restrict__ y, int S)
{
  int b = blockIdx.x;
  int lane = threadIdx.x;           // 0..31
  int i = lane & 15;
  bool act = lane < 16;
  float arow[16], vrow[16];
#pragma unroll
  for (int j = 0; j < 16; ++j) { arow[j] = A[i * 16 + j]; vrow[j] = Ainv[i * 16 + j]; }
  const float* bp = Bm + (size_t)b * S * 16;
  const float* cp = Cm + (size_t)b * S * 16;
  const float* dp = Dr + (size_t)b * S * 16;
  float h = 0.f;
  for (int s = 0; s < S; ++s) {
    float bv = bp[s * 16 + i];
    float cv = cp[s * 16 + i];
    float dv = dp[s * 16 + i];
    float delta = (dv > 20.f) ? dv : log1pf(__expf(dv));   // softplus
    float wv = 0.f;                                        // w = A^-1 B
#pragma unroll
    for (int j = 0; j < 16; ++j) wv += vrow[j] * __shfl(bv, j, 32);
    float hw = h + wv;
    float hn = 0.f;                  // h' = A_d (h + w) - w
#pragma unroll
    for (int j = 0; j < 16; ++j) hn += __expf(delta * arow[j]) * __shfl(hw, j, 32);
    hn -= wv;
    h = act ? hn : 0.f;
    float part = act ? (cv * h) : 0.f;
#pragma unroll
    for (int off = 8; off > 0; off >>= 1) part += __shfl_xor(part, off, 32);
    if (lane == 0) y[(size_t)b * S + s] = part;
  }
}

// ---------------------------------------------------------------- y-broadcast * gate -> bf16
__global__ void gate_mul_kernel(const u16* __restrict__ gbf, const float* __restrict__ y,
                                u16* __restrict__ g2, size_t total, int di)
{
  size_t i  = (size_t)blockIdx.x * blockDim.x + threadIdx.x;
  size_t st = (size_t)gridDim.x * blockDim.x;
  for (; i < total; i += st) {
    size_t row = i / (size_t)di;
    g2[i] = f2bf(bf2f(gbf[i]) * y[row]);
  }
}

// ---------------------------------------------------------------- residual + LayerNorm
__global__ __launch_bounds__(256)
void ln_kernel(const float* __restrict__ yr, const float* __restrict__ x,
               const float* __restrict__ g, const float* __restrict__ bta,
               float* __restrict__ out, int D)
{
  __shared__ float s1[256], s2[256];
  int row = blockIdx.x;
  const float* yp = yr + (size_t)row * D;
  const float* xp = x  + (size_t)row * D;
  float s = 0.f, ss = 0.f;
  for (int c = threadIdx.x; c < D; c += 256) {
    float z = yp[c] + xp[c];
    s += z; ss += z * z;
  }
  s1[threadIdx.x] = s; s2[threadIdx.x] = ss;
  __syncthreads();
  for (int off = 128; off > 0; off >>= 1) {
    if ((int)threadIdx.x < off) { s1[threadIdx.x] += s1[threadIdx.x + off]; s2[threadIdx.x] += s2[threadIdx.x + off]; }
    __syncthreads();
  }
  float mu  = s1[0] / (float)D;
  float var = s2[0] / (float)D - mu * mu;
  float inv = rsqrtf(var + 1e-5f);
  for (int c = threadIdx.x; c < D; c += 256) {
    float z = yp[c] + xp[c];
    out[(size_t)row * D + c] = (z - mu) * inv * g[c] + bta[c];
  }
}

// ---------------------------------------------------------------- launcher
extern "C" void kernel_launch(void* const* d_in, const int* in_sizes, int n_in,
                              void* d_out, int out_size, void* d_ws, size_t ws_size,
                              hipStream_t stream)
{
  (void)in_sizes; (void)n_in; (void)out_size; (void)ws_size;
  const float* x    = (const float*)d_in[0];
  const float* w1   = (const float*)d_in[1];
  const float* b1   = (const float*)d_in[2];
  const float* Bw   = (const float*)d_in[3];
  const float* Bb   = (const float*)d_in[4];
  const float* Cw   = (const float*)d_in[5];
  const float* Cb   = (const float*)d_in[6];
  const float* Dw   = (const float*)d_in[7];
  const float* Db   = (const float*)d_in[8];
  const float* Amat = (const float*)d_in[9];
  const float* Ow   = (const float*)d_in[10];
  const float* Ob   = (const float*)d_in[11];
  const float* lng  = (const float*)d_in[12];
  const float* lnb  = (const float*)d_in[13];
  float* out = (float*)d_out;

  const int Bt = 4, S = 2048, D = 1024, DI = 2048, N2 = 4096, ST = 16;
  const int Mr = Bt * S;   // 8192

  char* wp = (char*)d_ws;
  auto alloc = [&](size_t bytes) -> void* {
    void* p = (void*)wp;
    wp += (bytes + 255) & ~(size_t)255;
    return p;
  };
  u16*   xbf  = (u16*)alloc((size_t)Mr * D  * 2);
  u16*   w1bf = (u16*)alloc((size_t)N2 * D  * 2);
  u16*   bwbf = (u16*)alloc((size_t)ST * DI * 2);
  u16*   cwbf = (u16*)alloc((size_t)ST * DI * 2);
  u16*   dwbf = (u16*)alloc((size_t)ST * DI * 2);
  u16*   owbf = (u16*)alloc((size_t)D  * DI * 2);
  u16*   ubf  = (u16*)alloc((size_t)Mr * DI * 2);
  u16*   gbf  = (u16*)alloc((size_t)Mr * DI * 2);
  float* Bm   = (float*)alloc((size_t)Mr * ST * 4);
  float* Cmx  = (float*)alloc((size_t)Mr * ST * 4);
  float* Dr   = (float*)alloc((size_t)Mr * ST * 4);
  float* Ainv = (float*)alloc((size_t)ST * ST * 4);
  float* yv   = (float*)alloc((size_t)Mr * 4);
  u16*   g2bf = (u16*)alloc((size_t)Mr * DI * 2);
  float* o1   = (float*)alloc((size_t)Mr * D * 4);

  // 1) precision conversion (f32 -> bf16)
  cvt_bf16_kernel<<<4096, 256, 0, stream>>>(x,  xbf,  (size_t)Mr * D);
  cvt_bf16_kernel<<<4096, 256, 0, stream>>>(w1, w1bf, (size_t)N2 * D);
  cvt_bf16_kernel<<<128,  256, 0, stream>>>(Bw, bwbf, (size_t)ST * DI);
  cvt_bf16_kernel<<<128,  256, 0, stream>>>(Cw, cwbf, (size_t)ST * DI);
  cvt_bf16_kernel<<<128,  256, 0, stream>>>(Dw, dwbf, (size_t)ST * DI);
  cvt_bf16_kernel<<<4096, 256, 0, stream>>>(Ow, owbf, (size_t)D * DI);

  // 2) A^-1
  invertA_kernel<<<1, 32, 0, stream>>>(Amat, Ainv);

  // 3) in_proj GEMM with fused split+SiLU epilogue -> u(bf16), gate(bf16)
  dim3 g1((N2 + BN - 1) / BN, (Mr + BM - 1) / BM);
  gemm_bf16_wmma<<<g1, 256, 0, stream>>>(xbf, w1bf, b1, nullptr, ubf, gbf, Mr, N2, D, 1, DI);

  // 4) B/C/delta projections (N = 16)
  dim3 gs((ST + BN - 1) / BN, (Mr + BM - 1) / BM);
  gemm_bf16_wmma<<<gs, 256, 0, stream>>>(ubf, bwbf, Bb, Bm,  nullptr, nullptr, Mr, ST, DI, 0, 0);
  gemm_bf16_wmma<<<gs, 256, 0, stream>>>(ubf, cwbf, Cb, Cmx, nullptr, nullptr, Mr, ST, DI, 0, 0);
  gemm_bf16_wmma<<<gs, 256, 0, stream>>>(ubf, dwbf, Db, Dr,  nullptr, nullptr, Mr, ST, DI, 0, 0);

  // 5) sequential selective scan, one wave per batch
  scan_kernel<<<Bt, 32, 0, stream>>>(Bm, Cmx, Dr, Amat, Ainv, yv, S);

  // 6) y * gate -> bf16
  gate_mul_kernel<<<4096, 256, 0, stream>>>(gbf, yv, g2bf, (size_t)Mr * DI, DI);

  // 7) out_proj GEMM
  dim3 g3((D + BN - 1) / BN, (Mr + BM - 1) / BM);
  gemm_bf16_wmma<<<g3, 256, 0, stream>>>(g2bf, owbf, Ob, o1, nullptr, nullptr, Mr, D, DI, 0, 0);

  // 8) residual + LayerNorm
  ln_kernel<<<Mr, 256, 0, stream>>>(o1, x, lng, lnb, out, D);
}
